// QKV_19138374271378
// MI455X (gfx1250) — matrix operand order repaired and checked by reading, not compiled, not run on hardware
//
#include <hip/hip_runtime.h>
#include <math.h>

// ---------------------------------------------------------------------------
// Graph-attention pipeline for MI455X (gfx1250, wave32, WMMA 16x16x32 f16).
// Memory bound (~2.1 GB HBM traffic/call): h_E read once via async LDS
// staging; K consumed in LDS (never hits HBM); weights pre-packed once into
// per-lane f16 WMMA B-fragments so the hot loop does vector loads only.
// ---------------------------------------------------------------------------

#define N_NODES 50000
#define N_EDGES 800000
#define HDIM 128
#define DIN 384
#define NHEADS 4

typedef __attribute__((ext_vector_type(16))) _Float16 v16h;
typedef __attribute__((ext_vector_type(8)))  float    v8f;
typedef int v4i __attribute__((vector_size(16)));

#if defined(__has_builtin)
# if __has_builtin(__builtin_amdgcn_global_load_async_to_lds_b128)
#  define HAVE_ASYNC_LDS 1
# endif
#endif
#ifndef HAVE_ASYNC_LDS
# define HAVE_ASYNC_LDS 0
#endif

__device__ __forceinline__ int lane_id() { return (int)(threadIdx.x & 31); }

// ---- async global->LDS staging (CDNA5 ASYNCcnt path, with fallback) --------
__device__ __forceinline__ void stage_b128(const float* __restrict__ g, float* l) {
#if HAVE_ASYNC_LDS
  __builtin_amdgcn_global_load_async_to_lds_b128(
      (__attribute__((address_space(1))) v4i*)(g),
      (__attribute__((address_space(3))) v4i*)(l), 0, 0);
#else
  *(float4*)l = *(const float4*)g;
#endif
}
__device__ __forceinline__ void stage_wait() {
#if HAVE_ASYNC_LDS
# if __has_builtin(__builtin_amdgcn_s_wait_asynccnt)
  __builtin_amdgcn_s_wait_asynccnt(0);
# else
  asm volatile("s_wait_asynccnt 0" ::: "memory");
# endif
#endif
}

// ---- WMMA fragment helpers (CDNA5 wave32 layouts, CDNA5 ISA) ----
// 16-bit A 16x32 (MxK): lanes 0-15 -> M=lane, elems 0..7 = K 0..7, 8..15 = K 16..23
//                       lanes 16-31 -> M=lane-16, K 8..15 / 24..31
// Vectorized: 4x ds_load_b128 per lane from a 16x32 f32 row-major LDS tile.
__device__ __forceinline__ v16h a_frag_lds(const float* As) {
  const int lane = lane_id();
  const int half = lane >> 4;
  const int m    = lane & 15;
  const float4* p = (const float4*)(As + m * 32 + half * 8);
  const float4 x0 = p[0], x1 = p[1];   // K = half*8 + 0..7
  const float4 y0 = p[4], y1 = p[5];   // K = 16 + half*8 + 0..7
  v16h a;
  a[0]=(_Float16)x0.x;  a[1]=(_Float16)x0.y;  a[2]=(_Float16)x0.z;  a[3]=(_Float16)x0.w;
  a[4]=(_Float16)x1.x;  a[5]=(_Float16)x1.y;  a[6]=(_Float16)x1.z;  a[7]=(_Float16)x1.w;
  a[8]=(_Float16)y0.x;  a[9]=(_Float16)y0.y;  a[10]=(_Float16)y0.z; a[11]=(_Float16)y0.w;
  a[12]=(_Float16)y1.x; a[13]=(_Float16)y1.y; a[14]=(_Float16)y1.z; a[15]=(_Float16)y1.w;
  return a;
}

// Pre-packed B fragment: Wp[kt][nt][lane][16 f16] -> one 32-byte vector load.
__device__ __forceinline__ v16h b_frag_packed(const _Float16* __restrict__ Wp,
                                              int kt, int nt) {
  return *(const v16h*)(Wp + (((size_t)kt * 8 + nt) * 32 + lane_id()) * 16);
}

// f32 C/D 16x16: VGPR j -> (lanes 0-15: M=j, N=lane) (lanes 16-31: M=j+8, N=lane-16)
__device__ __forceinline__ void store_cd(float* __restrict__ C, int ldc,
                                         int row0, int col0, v8f acc) {
  const int lane = lane_id();
  const int half = lane >> 4;
  const int n    = lane & 15;
#pragma unroll
  for (int j = 0; j < 8; ++j)
    C[(size_t)(row0 + j + half * 8) * ldc + (col0 + n)] = acc[j];
}

__device__ __forceinline__ v8f wmma16(v16h a, v16h b, v8f c) {
  return __builtin_amdgcn_wmma_f32_16x16x32_f16(false, a, false, b,
                                                (short)0, c, false, false);
}

// ---- order-preserving uint encoding of float (for atomic float max) --------
__device__ __forceinline__ unsigned enc_f32(float f) {
  unsigned u = __float_as_uint(f);
  return (u & 0x80000000u) ? ~u : (u | 0x80000000u);
}
__device__ __forceinline__ float dec_f32(unsigned k) {
  unsigned u = (k & 0x80000000u) ? (k ^ 0x80000000u) : ~k;
  return __uint_as_float(u);
}

// ---------------------------------------------------------------------------
// Pack W[Kdim x 128] (f32, row-major) into per-lane f16 B fragments:
// Wp[kt][nt][lane][e] = (f16) W[(kt*32 + (lane>>4)*8 + (e&7) + (e>>3)*16) * 128 + nt*16 + (lane&15)]
// ---------------------------------------------------------------------------
__global__ void k_pack_w(const float* __restrict__ W, _Float16* __restrict__ Wp,
                         int ktiles) {
  const int idx = blockIdx.x * 256 + (int)threadIdx.x;
  if (idx >= ktiles * 4096) return;
  const int e    = idx & 15;
  const int lane = (idx >> 4) & 31;
  const int nt   = (idx >> 9) & 7;
  const int kt   = idx >> 12;
  const int k = (lane >> 4) * 8 + (e & 7) + (e >> 3) * 16;
  const int n = lane & 15;
  Wp[idx] = (_Float16)W[(size_t)(kt * 32 + k) * HDIM + nt * 16 + n];
}

// ---------------------------------------------------------------------------
// C[M x 128] = A[M x Kdim] * W (pre-packed).  16 rows/block, 4 waves, each
// wave owns a 32-column slab.  A tile staged to LDS via async copy.
// ---------------------------------------------------------------------------
__global__ void k_gemm_n128(const float* __restrict__ A,
                            const _Float16* __restrict__ Wp,
                            float* __restrict__ C, int Kdim) {
  __shared__ float As[16 * 32];
  const int row0 = blockIdx.x * 16;
  const int t    = (int)threadIdx.x;
  const int wave = t >> 5;
  const int col0 = wave * 32;
  v8f acc0 = {}, acc1 = {};
  for (int k0 = 0; k0 < Kdim; k0 += 32) {
    __syncthreads();
    const float* gsrc = A + (size_t)(row0 + (t >> 3)) * Kdim + k0 + (t & 7) * 4;
    stage_b128(gsrc, As + t * 4);
    stage_wait();
    __syncthreads();
    v16h a  = a_frag_lds(As);
    const int kt = k0 >> 5;
    v16h b0 = b_frag_packed(Wp, kt, wave * 2);
    v16h b1 = b_frag_packed(Wp, kt, wave * 2 + 1);
    acc0 = wmma16(a, b0, acc0);
    acc1 = wmma16(a, b1, acc1);
  }
  store_cd(C, HDIM, row0, col0, acc0);
  store_cd(C, HDIM, row0, col0 + 16, acc1);
}

// ---------------------------------------------------------------------------
// Fused K/V GEMM + per-edge logits.  16 edges/block; h_E read exactly once
// (async-staged to LDS); K parked in LDS and consumed for logits locally.
// ---------------------------------------------------------------------------
__global__ void k_kv_logits(const float* __restrict__ hE,
                            const _Float16* __restrict__ WpK,
                            const _Float16* __restrict__ WpV,
                            const float* __restrict__ Q,
                            const int*   __restrict__ center,
                            float* __restrict__ V,
                            float* __restrict__ logits) {
  __shared__ float As[16 * 32];     // h_E tile (16 edges x 32 k)
  __shared__ float Ks[16 * HDIM];   // K tile   (16 edges x 128)

  const int e0   = blockIdx.x * 16;
  const int t    = (int)threadIdx.x;
  const int wave = t >> 5;
  const int col0 = wave * 32;

  v8f accK0 = {}, accK1 = {}, accV0 = {}, accV1 = {};

  for (int k0 = 0; k0 < DIN; k0 += 32) {
    __syncthreads();   // previous-iteration As readers done
    const float* gsrc = hE + (size_t)(e0 + (t >> 3)) * DIN + k0 + (t & 7) * 4;
    stage_b128(gsrc, As + t * 4);
    if (k0 + 32 < DIN) __builtin_prefetch(gsrc + 32, 0, 1);  // next k-tile
    stage_wait();
    __syncthreads();

    v16h a = a_frag_lds(As);
    const int kt = k0 >> 5;
    v16h bk0 = b_frag_packed(WpK, kt, wave * 2);
    v16h bk1 = b_frag_packed(WpK, kt, wave * 2 + 1);
    v16h bv0 = b_frag_packed(WpV, kt, wave * 2);
    v16h bv1 = b_frag_packed(WpV, kt, wave * 2 + 1);
    accK0 = wmma16(a, bk0, accK0);
    accK1 = wmma16(a, bk1, accK1);
    accV0 = wmma16(a, bv0, accV0);
    accV1 = wmma16(a, bv1, accV1);
  }

  // V -> global (needed by scatter phase)
  store_cd(V, HDIM, e0, col0, accV0);
  store_cd(V, HDIM, e0, col0 + 16, accV1);

  // K -> LDS (consumed locally, never hits HBM)
  {
    const int lane = t & 31;
    const int half = lane >> 4;
    const int n    = lane & 15;
#pragma unroll
    for (int j = 0; j < 8; ++j) {
      Ks[(j + half * 8) * HDIM + col0 + n]      = accK0[j];
      Ks[(j + half * 8) * HDIM + col0 + 16 + n] = accK1[j];
    }
  }
  __syncthreads();

  // logits[e, h] = <Q[center[e], h, :], K[e, h, :]> / sqrt(32)
  if (t < 64) {
    const int el = t >> 2;
    const int h  = t & 3;
    const int e  = e0 + el;
    const int c  = center[e];
    const float* q  = Q  + (size_t)c * HDIM + h * 32;
    const float* kk = Ks + el * HDIM + h * 32;
    float acc = 0.0f;
#pragma unroll
    for (int i = 0; i < 32; ++i) acc += q[i] * kk[i];
    logits[(size_t)e * NHEADS + h] = acc * 0.17677669529663689f; // 1/sqrt(32)
  }
}

// ---------------------------------------------------------------------------
// Segment softmax pieces (smax/denom/h_att are L2-resident)
// ---------------------------------------------------------------------------
__global__ void k_segmax(const float* __restrict__ logits,
                         const int* __restrict__ center,
                         unsigned* __restrict__ smax) {
  const int idx = blockIdx.x * blockDim.x + (int)threadIdx.x;
  if (idx >= N_EDGES * NHEADS) return;
  const int e = idx >> 2, h = idx & 3;
  const int c = center[e];
  atomicMax(&smax[c * NHEADS + h], enc_f32(logits[idx]));
}

__global__ void k_exp_denom(float* __restrict__ logits,   // in: logits, out: ex
                            const int* __restrict__ center,
                            const unsigned* __restrict__ smax,
                            float* __restrict__ denom) {
  const int idx = blockIdx.x * blockDim.x + (int)threadIdx.x;
  if (idx >= N_EDGES * NHEADS) return;
  const int e = idx >> 2, h = idx & 3;
  const int c = center[e];
  const float m  = dec_f32(smax[c * NHEADS + h]);
  const float ex = __expf(logits[idx] - m);
  logits[idx] = ex;
  atomicAdd(&denom[c * NHEADS + h], ex);
}

__global__ void k_scatter(const float* __restrict__ ex,
                          const float* __restrict__ denom,
                          const int* __restrict__ center,
                          const float* __restrict__ V,
                          float* __restrict__ h_att) {
  const int t = (int)threadIdx.x;
  const int e = blockIdx.x * 4 + (t >> 5);
  const int l = t & 31;
  const int col = l * 4;
  const int h = col >> 5;
  const int c = center[e];
  const float att = ex[(size_t)e * NHEADS + h] / denom[c * NHEADS + h];
  const float4 v = *(const float4*)(V + (size_t)e * HDIM + col);
  float* dst = h_att + (size_t)c * HDIM + col;
  atomicAdd(dst + 0, att * v.x);
  atomicAdd(dst + 1, att * v.y);
  atomicAdd(dst + 2, att * v.z);
  atomicAdd(dst + 3, att * v.w);
}

// ---------------------------------------------------------------------------
extern "C" void kernel_launch(void* const* d_in, const int* in_sizes, int n_in,
                              void* d_out, int out_size, void* d_ws, size_t ws_size,
                              hipStream_t stream) {
  const float* h_V    = (const float*)d_in[0];
  const float* h_E    = (const float*)d_in[1];
  const int*   center = (const int*)d_in[2];
  // d_in[3] = batch_id (unused by reference)
  const float* W_Q    = (const float*)d_in[4];
  const float* W_K    = (const float*)d_in[5];
  const float* W_V    = (const float*)d_in[6];
  const float* W_O    = (const float*)d_in[7];
  float* out = (float*)d_out;

  // Workspace carve-up (~476 MB)
  float*    Q      = (float*)d_ws;                                   // 6.4M f
  float*    V      = Q + (size_t)N_NODES * HDIM;                     // 102.4M f
  float*    logits = V + (size_t)N_EDGES * HDIM;                     // 3.2M f
  unsigned* smax   = (unsigned*)(logits + (size_t)N_EDGES * NHEADS); // 0.2M u32
  float*    denom  = (float*)(smax + (size_t)N_NODES * NHEADS);      // 0.2M f
  float*    h_att  = denom + (size_t)N_NODES * NHEADS;               // 6.4M f
  _Float16* WpQ    = (_Float16*)(h_att + (size_t)N_NODES * HDIM);    // 16K f16
  _Float16* WpK    = WpQ + 4  * 4096;                                // 48K f16
  _Float16* WpV    = WpK + 12 * 4096;                                // 48K f16
  _Float16* WpO    = WpV + 12 * 4096;                                // 16K f16

  // Zero [smax | denom | h_att]: 0-bits == 0.0f == min of max-encoding.
  const size_t zero_words = (size_t)N_NODES * NHEADS * 2 + (size_t)N_NODES * HDIM;
  (void)hipMemsetAsync(smax, 0, zero_words * sizeof(float), stream);

  // 0) pre-pack weights into WMMA B-fragment layout (f16), once per call
  k_pack_w<<<(4  * 4096) / 256, 256, 0, stream>>>(W_Q, WpQ, 4);
  k_pack_w<<<(12 * 4096) / 256, 256, 0, stream>>>(W_K, WpK, 12);
  k_pack_w<<<(12 * 4096) / 256, 256, 0, stream>>>(W_V, WpV, 12);
  k_pack_w<<<(4  * 4096) / 256, 256, 0, stream>>>(W_O, WpO, 4);

  // 1) Q = h_V @ W_Q
  k_gemm_n128<<<N_NODES / 16, 128, 0, stream>>>(h_V, WpQ, Q, HDIM);

  // 2) fused K/V GEMM + logits (K stays in LDS)
  k_kv_logits<<<N_EDGES / 16, 128, 0, stream>>>(h_E, WpK, WpV, Q, center, V, logits);

  // 3) segment max
  const int eh = N_EDGES * NHEADS;
  k_segmax<<<(eh + 255) / 256, 256, 0, stream>>>(logits, center, smax);

  // 4) exp + denom
  k_exp_denom<<<(eh + 255) / 256, 256, 0, stream>>>(logits, center, smax, denom);

  // 5) scatter attend * V into nodes
  k_scatter<<<N_EDGES / 4, 128, 0, stream>>>(logits, denom, center, V, h_att);

  // 6) out = h_att @ W_O
  k_gemm_n128<<<N_NODES / 16, 128, 0, stream>>>(h_att, WpO, out, HDIM);
}
